// PositionalGNNLayer_53652731462301
// MI455X (gfx1250) — matrix operand order repaired and checked by reading, not compiled, not run on hardware
//
#include <hip/hip_runtime.h>

#define N_NODES 8192
#define IN_DIM  256
#define HID     128
#define LN_EPS  1e-5f

typedef __attribute__((ext_vector_type(16))) __bf16 v16bf;
typedef __attribute__((ext_vector_type(8)))  float  v8f;

union Frag {
    v16bf v;
    unsigned u[8];
    float4 f4[2];
};

__device__ __forceinline__ unsigned short f2bf(float f) {
    unsigned u = __float_as_uint(f);
    u += 0x7FFFu + ((u >> 16) & 1u);   // round-to-nearest-even
    return (unsigned short)(u >> 16);
}

__device__ __forceinline__ v8f zero8() {
    v8f z;
    #pragma unroll
    for (int e = 0; e < 8; e++) z[e] = 0.f;
    return z;
}

// ---------------------------------------------------------------------------
// Kernel 1: messages = node_features @ msg_w + msg_b
//   writes fp32 messages and bf16 transposed messages (for WMMA B-fragments)
// ---------------------------------------------------------------------------
__global__ __launch_bounds__(128) void msg_kernel(
    const float* __restrict__ nf, const float* __restrict__ mw,
    const float* __restrict__ mb,
    float* __restrict__ msg, unsigned short* __restrict__ msgT)
{
    const int i = blockIdx.x;
    const int h = threadIdx.x;
    const float* nfi = nf + (size_t)i * IN_DIM;
    float acc = mb[h];
    #pragma unroll 4
    for (int d = 0; d < IN_DIM; d++)
        acc = fmaf(nfi[d], mw[d * HID + h], acc);
    msg[(size_t)i * HID + h] = acc;
    msgT[(size_t)h * N_NODES + i] = f2bf(acc);
}

// ---------------------------------------------------------------------------
// Kernel 2: q = scale * messages @ gate_q_w ; k = messages @ gate_k_w  (bf16)
// ---------------------------------------------------------------------------
__global__ __launch_bounds__(128) void qk_kernel(
    const float* __restrict__ msg, const float* __restrict__ qw,
    const float* __restrict__ kw,
    unsigned short* __restrict__ qb, unsigned short* __restrict__ kb)
{
    const int i = blockIdx.x;
    const int h = threadIdx.x;
    const float* mi = msg + (size_t)i * HID;
    float aq = 0.f, ak = 0.f;
    #pragma unroll 4
    for (int d = 0; d < HID; d++) {
        float m = mi[d];
        aq = fmaf(m, qw[d * HID + h], aq);
        ak = fmaf(m, kw[d * HID + h], ak);
    }
    const float scale = 0.08838834764831845f;  // 1/sqrt(128), tau = 1
    qb[(size_t)i * HID + h] = f2bf(aq * scale);
    kb[(size_t)i * HID + h] = f2bf(ak);
}

// ---------------------------------------------------------------------------
// Kernel 3: fused  scores -> sigmoid gate * adj -> aggregate, all in registers.
// One wave (32 lanes) owns a 16-row output tile; loops over 32-col chunks.
// First GEMM computed transposed (A = k rows, B = q rows) so the D-tile layout
// is directly the A-fragment layout of the second GEMM after f32->bf16 pack.
// ---------------------------------------------------------------------------
__global__ __launch_bounds__(128) void fused_gate_agg(
    const float* __restrict__ adj,
    const unsigned short* __restrict__ qb,
    const unsigned short* __restrict__ kb,
    const unsigned short* __restrict__ mT,
    const float* __restrict__ gbias_p,
    float* __restrict__ agg)
{
    const int lane = threadIdx.x & 31;
    const int tile = blockIdx.x * 4 + (threadIdx.x >> 5);
    const int i0 = tile << 4;
    const int lm = lane & 15;
    const int hi = lane >> 4;
    const float gb = gbias_p[0];

    // q B-fragments (K = message dim, 4 chunks of 32) -- resident all loop
    Frag qf[4];
    {
        const unsigned short* qrow = qb + (size_t)(i0 + lm) * HID + hi * 8;
        #pragma unroll
        for (int t = 0; t < 4; t++) {
            qf[t].f4[0] = *(const float4*)(qrow + t * 32);
            qf[t].f4[1] = *(const float4*)(qrow + t * 32 + 16);
        }
    }

    v8f aggv[8];
    #pragma unroll
    for (int h = 0; h < 8; h++) aggv[h] = zero8();

    for (int j0 = 0; j0 < N_NODES; j0 += 32) {
        // ---- scores^T tiles: D[M=j, N=i] = sum_d k[j][d] * q[i][d]
        v8f s0 = zero8();
        v8f s1 = zero8();
        const unsigned short* krow0 = kb + (size_t)(j0 + lm) * HID + hi * 8;
        const unsigned short* krow1 = krow0 + 16 * HID;
        #pragma unroll
        for (int t = 0; t < 4; t++) {
            Frag ka;
            ka.f4[0] = *(const float4*)(krow0 + t * 32);
            ka.f4[1] = *(const float4*)(krow0 + t * 32 + 16);
            s0 = __builtin_amdgcn_wmma_f32_16x16x32_bf16(
                     false, ka.v, false, qf[t].v, (short)0, s0, false, false);
        }
        #pragma unroll
        for (int t = 0; t < 4; t++) {
            Frag ka;
            ka.f4[0] = *(const float4*)(krow1 + t * 32);
            ka.f4[1] = *(const float4*)(krow1 + t * 32 + 16);
            s1 = __builtin_amdgcn_wmma_f32_16x16x32_bf16(
                     false, ka.v, false, qf[t].v, (short)0, s1, false, false);
        }

        // ---- adjacency (this lane sees row i0+lm, cols j0 + 8*hi + r)
        const float* arow = adj + (size_t)(i0 + lm) * N_NODES + j0 + hi * 8;
        __builtin_prefetch((const void*)(arow + 256), 0, 0);
        float a[8], b[8];
        *(float4*)&a[0] = *(const float4*)(arow);
        *(float4*)&a[4] = *(const float4*)(arow + 4);
        *(float4*)&b[0] = *(const float4*)(arow + 16);
        *(float4*)&b[4] = *(const float4*)(arow + 20);

        // ---- gate = sigmoid(score + bias) * adj, pack f32 pairs -> bf16 A-frag
        Frag gA;
        #pragma unroll
        for (int p = 0; p < 4; p++) {
            float x0 = s0[2 * p]     + gb;
            float x1 = s0[2 * p + 1] + gb;
            float g0 = a[2 * p]     / (1.f + __expf(-x0));
            float g1 = a[2 * p + 1] / (1.f + __expf(-x1));
            gA.u[p] = (unsigned)f2bf(g0) | ((unsigned)f2bf(g1) << 16);
        }
        #pragma unroll
        for (int p = 0; p < 4; p++) {
            float x0 = s1[2 * p]     + gb;
            float x1 = s1[2 * p + 1] + gb;
            float g0 = b[2 * p]     / (1.f + __expf(-x0));
            float g1 = b[2 * p + 1] / (1.f + __expf(-x1));
            gA.u[4 + p] = (unsigned)f2bf(g0) | ((unsigned)f2bf(g1) << 16);
        }

        // ---- aggregated += gates @ messages (8 hidden groups of 16)
        #pragma unroll
        for (int h = 0; h < 8; h++) {
            Frag mB;
            const unsigned short* mrow =
                mT + (size_t)(16 * h + lm) * N_NODES + j0 + hi * 8;
            mB.f4[0] = *(const float4*)(mrow);
            mB.f4[1] = *(const float4*)(mrow + 16);
            aggv[h] = __builtin_amdgcn_wmma_f32_16x16x32_bf16(
                          false, gA.v, false, mB.v, (short)0, aggv[h], false, false);
        }
    }

    // ---- store aggregated tile (D layout: lane = N column, VGPR r = row)
    #pragma unroll
    for (int h = 0; h < 8; h++) {
        #pragma unroll
        for (int r = 0; r < 8; r++) {
            agg[(size_t)(i0 + r + 8 * hi) * HID + 16 * h + lm] = aggv[h][r];
        }
    }
}

// ---------------------------------------------------------------------------
// Kernel 4: updated = relu([nf, agg] @ upd_w + upd_b); LayerNorm over HID.
// One 128-thread block per row.
// ---------------------------------------------------------------------------
__global__ __launch_bounds__(128) void update_ln(
    const float* __restrict__ nf, const float* __restrict__ agg,
    const float* __restrict__ uw, const float* __restrict__ ub,
    const float* __restrict__ gamma, const float* __restrict__ beta,
    float* __restrict__ out)
{
    const int i = blockIdx.x;
    const int h = threadIdx.x;
    const float* nfi = nf + (size_t)i * IN_DIM;
    const float* agi = agg + (size_t)i * HID;
    float acc = ub[h];
    #pragma unroll 4
    for (int d = 0; d < IN_DIM; d++)
        acc = fmaf(nfi[d], uw[d * HID + h], acc);
    #pragma unroll 4
    for (int d = 0; d < HID; d++)
        acc = fmaf(agi[d], uw[(IN_DIM + d) * HID + h], acc);
    float r = fmaxf(acc, 0.f);

    __shared__ float s1[128];
    __shared__ float s2[128];
    s1[h] = r;
    s2[h] = r * r;
    __syncthreads();
    for (int off = 64; off > 0; off >>= 1) {
        if (h < off) { s1[h] += s1[h + off]; s2[h] += s2[h + off]; }
        __syncthreads();
    }
    float mu  = s1[0] * (1.f / HID);
    float var = s2[0] * (1.f / HID) - mu * mu;
    float inv = rsqrtf(var + LN_EPS);
    out[(size_t)i * HID + h] = (r - mu) * inv * gamma[h] + beta[h];
}

// ---------------------------------------------------------------------------
extern "C" void kernel_launch(void* const* d_in, const int* in_sizes, int n_in,
                              void* d_out, int out_size, void* d_ws, size_t ws_size,
                              hipStream_t stream) {
    const float* nf    = (const float*)d_in[0];
    const float* adj   = (const float*)d_in[1];
    const float* msg_w = (const float*)d_in[2];
    const float* msg_b = (const float*)d_in[3];
    const float* upd_w = (const float*)d_in[4];
    const float* upd_b = (const float*)d_in[5];
    const float* qw    = (const float*)d_in[6];
    const float* kw    = (const float*)d_in[7];
    const float* gbias = (const float*)d_in[8];
    const float* gam   = (const float*)d_in[9];
    const float* bet   = (const float*)d_in[10];
    float* out = (float*)d_out;

    char* ws = (char*)d_ws;
    float*          msg_f32 = (float*)(ws);                               // 4 MB
    unsigned short* msgT    = (unsigned short*)(ws + (((size_t)4)  << 20)); // 2 MB
    unsigned short* qbuf    = (unsigned short*)(ws + (((size_t)6)  << 20)); // 2 MB
    unsigned short* kbuf    = (unsigned short*)(ws + (((size_t)8)  << 20)); // 2 MB
    float*          aggbuf  = (float*)(ws + (((size_t)10) << 20));          // 4 MB

    msg_kernel<<<N_NODES, 128, 0, stream>>>(nf, msg_w, msg_b, msg_f32, msgT);
    qk_kernel<<<N_NODES, 128, 0, stream>>>(msg_f32, qw, kw, qbuf, kbuf);
    fused_gate_agg<<<N_NODES / 64, 128, 0, stream>>>(adj, qbuf, kbuf, msgT,
                                                     gbias, aggbuf);
    update_ln<<<N_NODES, 128, 0, stream>>>(nf, aggbuf, upd_w, upd_b, gam, bet, out);
}
